// hand_lstm_6408091206365
// MI455X (gfx1250) — matrix operand order repaired and checked by reading, not compiled, not run on hardware
//
#include <hip/hip_runtime.h>

// ---------------------------------------------------------------------------
// MI455X (gfx1250) fused 2-layer LSTM + linear heads.
//
// - Recurrence is latency-bound (33 MFLOP/step, sequential over T=512), so we
//   run one persistent workgroup per 32 batch rows (rows are independent),
//   c-state in registers, h-state in LDS, bf16 weight B-fragments hoisted into
//   VGPRs for the whole scan. All matmuls: v_wmma_f32_16x16x32_bf16.
// - Per-step 8KB tile movement (h writeback, layer-1 input staging) uses the
//   Tensor Data Mover (tensor_load_to_lds / tensor_store_from_lds, TENSORcnt),
//   with the layer-1 input double-buffered so the TDM load for step t+1
//   overlaps the WMMA work of step t.
// - The two linear heads are fused into one [B*T,128]x[128,63] WMMA GEMM
//   (bandwidth-bound: ~67MB @ 23.3 TB/s).
// ---------------------------------------------------------------------------

typedef __bf16 bf16;
typedef bf16  v16bf __attribute__((ext_vector_type(16)));
typedef float v8f   __attribute__((ext_vector_type(8)));
typedef unsigned int u32;
typedef u32 u32x4 __attribute__((ext_vector_type(4)));
typedef u32 u32x8 __attribute__((ext_vector_type(8)));

#define Hh   128   // hidden
#define Tt   512   // timesteps
#define Bb   256   // batch
#define Ii   30    // input features (layer 0)
#define BT   32    // batch rows per workgroup
#define NTH  256   // threads per LSTM block (8 waves)

// Branch-free activations: v_exp_f32 + v_rcp_f32, saturate correctly at +-inf.
__device__ __forceinline__ float sigm_(float x) {
  return __builtin_amdgcn_rcpf(1.0f + __expf(-x));
}
__device__ __forceinline__ float tanh_(float x) {
  return 1.0f - 2.0f * __builtin_amdgcn_rcpf(__expf(2.0f * x) + 1.0f);
}

__device__ __forceinline__ v8f wmma_bf16(v16bf a, v16bf b, v8f c) {
  return __builtin_amdgcn_wmma_f32_16x16x32_bf16(false, a, false, b, (short)0, c, false, false);
}

// ---------------------------------------------------------------------------
// Tensor Data Mover: 1-D tile (nelem8 x 8 bytes) between LDS and global.
// D# group0: count=1, lds_addr, 57b global_addr, type=2.
// D# group1: data_size=8B, tensor_dim0=tile_dim0=nelem8, tensor_dim1=tile_dim1=1.
// ---------------------------------------------------------------------------
__device__ __forceinline__ void tdm_desc(u32x4& g0, u32x8& g1, u32 lds_off,
                                         const void* gptr, u32 nelem8) {
  unsigned long long ga = (unsigned long long)gptr;
  g0[0] = 1u;                                             // count=1, no gather
  g0[1] = lds_off;                                        // lds_addr (bytes)
  g0[2] = (u32)ga;                                        // global_addr[31:0]
  g0[3] = ((u32)(ga >> 32) & 0x01FFFFFFu) | 0x80000000u;  // addr[56:32] | type=2
  g1[0] = 3u << 16;                                       // data_size = 8B
  g1[1] = (nelem8 & 0xFFFFu) << 16;                       // tensor_dim0[15:0]
  g1[2] = ((nelem8 >> 16) & 0xFFFFu) | (1u << 16);        // dim0[31:16] | tensor_dim1=1
  g1[3] = (nelem8 & 0xFFFFu) << 16;                       // tile_dim0
  g1[4] = 1u;                                             // tile_dim1=1, tile_dim2=0
  g1[5] = nelem8;                                         // tensor_dim0_stride[31:0]
  g1[6] = 0u;
  g1[7] = 0u;
}
__device__ __forceinline__ void tdm_load(u32 lds_off, const void* gptr, u32 nelem8) {
  u32x4 g0; u32x8 g1;
  tdm_desc(g0, g1, lds_off, gptr, nelem8);
  asm volatile("tensor_load_to_lds %0, %1" :: "s"(g0), "s"(g1) : "memory");
}
__device__ __forceinline__ void tdm_store(u32 lds_off, const void* gptr, u32 nelem8) {
  u32x4 g0; u32x8 g1;
  tdm_desc(g0, g1, lds_off, gptr, nelem8);
  asm volatile("tensor_store_from_lds %0, %1" :: "s"(g0), "s"(g1) : "memory");
}
__device__ __forceinline__ void wait_tensor0() {
  __builtin_amdgcn_s_wait_tensorcnt((short)0);
}

// A-fragment (16x32 bf16, MxK): lane holds row M=lane&15 of tile m, one
// contiguous 16-element K-run at kt*32 + 16*(lane>>4) (matches B fragments).
__device__ __forceinline__ v16bf loadA_bf16(const bf16* base, int stride, int m, int kt, int lane) {
  int row = m * 16 + (lane & 15);
  int kb  = kt * 32 + ((lane >> 4) & 1) * 16;
  const bf16* p = base + row * stride + kb;
  v16bf a;
  ((uint4*)&a)[0] = *(const uint4*)(p);
  ((uint4*)&a)[1] = *(const uint4*)(p + 8);
  return a;
}

// B-fragment (32x16 bf16, KxN), W stored [N][K]: lane = column N=nb+(lane&15),
// one contiguous 16-element K-run at kt*32 + 16*(lane>>4).
__device__ __forceinline__ v16bf loadB_bf16(const bf16* W, int ldk, int nb, int kt, int lane) {
  int col = nb + (lane & 15);
  int kb  = kt * 32 + ((lane >> 4) & 1) * 16;
  const bf16* p = W + col * ldk + kb;
  v16bf b;
  ((uint4*)&b)[0] = *(const uint4*)(p);
  ((uint4*)&b)[1] = *(const uint4*)(p + 8);
  return b;
}

// Same B-fragment sourced from f32 weights with zero-padding past kvalid.
__device__ __forceinline__ v16bf loadB_f32(const float* W, int ldk, int nb, int kt, int kvalid, int lane) {
  int col = nb + (lane & 15);
  int kb  = kt * 32 + ((lane >> 4) & 1) * 16;
  v16bf b;
#pragma unroll
  for (int e = 0; e < 16; ++e) {
    int k = kb + e;
    float v = (k < kvalid) ? W[(size_t)col * ldk + k] : 0.0f;
    b[e] = (bf16)v;
  }
  return b;
}

// ---------------------------------------------------------------------------
// LSTM layer 0: x[B,T,30](f32) -> h1[T,B,128](bf16).  grid=8, block=256.
// ---------------------------------------------------------------------------
__global__ __launch_bounds__(NTH, 1)
void lstm_layer0(const float* __restrict__ x, const float* __restrict__ Wih,
                 const float* __restrict__ Whh, const float* __restrict__ bih,
                 const float* __restrict__ bhh, bf16* __restrict__ hout) {
  __shared__ bf16 smem[BT * Hh + BT * 32];  // [hs | xs]
  bf16* hs = smem;            // LDS offset 0, 8KB
  bf16* xs = smem + BT * Hh;  // LDS offset 8192, 2KB

  const int tid = threadIdx.x, lane = tid & 31, wave = tid >> 5;
  const int gw = wave * 16;
  const int b0 = blockIdx.x * BT;

  v16bf Bh[4][4], Bx[4];
  float biasv[4];
#pragma unroll
  for (int g = 0; g < 4; ++g) {
    int nb = g * Hh + gw;
    Bx[g] = loadB_f32(Wih, Ii, nb, 0, Ii, lane);
#pragma unroll
    for (int kt = 0; kt < 4; ++kt) Bh[g][kt] = loadB_f32(Whh, Hh, nb, kt, Hh, lane);
    int n = nb + (lane & 15);
    biasv[g] = bih[n] + bhh[n];
  }

  for (int i = tid; i < BT * Hh; i += NTH) hs[i] = (bf16)0.0f;

  v8f cst[2], hv[2];
#pragma unroll
  for (int m = 0; m < 2; ++m)
#pragma unroll
    for (int e = 0; e < 8; ++e) { cst[m][e] = 0.0f; hv[m][e] = 0.0f; }

  __syncthreads();

#pragma unroll 1
  for (int t = 0; t < Tt; ++t) {
    if (wave == 0) wait_tensor0();  // prior TDM store of hs done before rewrite
    __syncthreads();

    if (t > 0) {  // publish h_{t-1}; waves write disjoint column slices
#pragma unroll
      for (int m = 0; m < 2; ++m)
#pragma unroll
        for (int e = 0; e < 8; ++e) {
          int row = m * 16 + e + ((lane >> 4) & 1) * 8;
          hs[row * Hh + gw + (lane & 15)] = (bf16)hv[m][e];
        }
    }
    {  // stage x_t -> LDS (bf16, K padded to 32)
      int r = tid >> 3, j = tid & 7;
      const float* xp = x + ((size_t)(b0 + r) * Tt + t) * Ii;
#pragma unroll
      for (int q = 0; q < 4; ++q) {
        int k = j * 4 + q;
        xs[r * 32 + k] = (k < Ii) ? (bf16)xp[k] : (bf16)0.0f;
      }
    }
    __syncthreads();

    if (wave == 0 && t > 0)  // async DMA h_{t-1} tile out (overlaps WMMA reads)
      tdm_store(0u, hout + ((size_t)(t - 1) * Bb + b0) * Hh, (BT * Hh * 2) / 8);

    v16bf Ax[2], Ah[2][4];
#pragma unroll
    for (int m = 0; m < 2; ++m) {
      Ax[m] = loadA_bf16(xs, 32, m, 0, lane);
#pragma unroll
      for (int kt = 0; kt < 4; ++kt) Ah[m][kt] = loadA_bf16(hs, Hh, m, kt, lane);
    }

    v8f gacc[4][2];
#pragma unroll
    for (int g = 0; g < 4; ++g)
#pragma unroll
      for (int m = 0; m < 2; ++m) {
        v8f acc;
#pragma unroll
        for (int e = 0; e < 8; ++e) acc[e] = biasv[g];
        acc = wmma_bf16(Ax[m], Bx[g], acc);            // x @ Wih^T  (K=32)
#pragma unroll
        for (int kt = 0; kt < 4; ++kt)
          acc = wmma_bf16(Ah[m][kt], Bh[g][kt], acc);  // h @ Whh^T  (K=128)
        gacc[g][m] = acc;
      }

#pragma unroll
    for (int m = 0; m < 2; ++m)
#pragma unroll
      for (int e = 0; e < 8; ++e) {
        float ig = sigm_(gacc[0][m][e]);
        float fg = sigm_(gacc[1][m][e]);
        float gg = tanh_(gacc[2][m][e]);
        float og = sigm_(gacc[3][m][e]);
        float c = fg * cst[m][e] + ig * gg;
        cst[m][e] = c;
        hv[m][e] = og * tanh_(c);
      }
  }

  // flush h_{T-1}
  if (wave == 0) wait_tensor0();
  __syncthreads();
#pragma unroll
  for (int m = 0; m < 2; ++m)
#pragma unroll
    for (int e = 0; e < 8; ++e) {
      int row = m * 16 + e + ((lane >> 4) & 1) * 8;
      hs[row * Hh + gw + (lane & 15)] = (bf16)hv[m][e];
    }
  __syncthreads();
  if (wave == 0) {
    tdm_store(0u, hout + ((size_t)(Tt - 1) * Bb + b0) * Hh, (BT * Hh * 2) / 8);
    wait_tensor0();
  }
}

// ---------------------------------------------------------------------------
// LSTM layer 1: h1[T,B,128](bf16) -> h2[T,B,128](bf16).  grid=8, block=256.
// Input tile double-buffered via TDM (load for t+1 overlaps compute of t).
// ---------------------------------------------------------------------------
__global__ __launch_bounds__(NTH, 1)
void lstm_layer1(const bf16* __restrict__ hin, const float* __restrict__ Wih,
                 const float* __restrict__ Whh, const float* __restrict__ bih,
                 const float* __restrict__ bhh, bf16* __restrict__ hout) {
  __shared__ bf16 smem[BT * Hh * 3];  // [hs | is0 | is1]
  bf16* hs = smem;                    // LDS offset 0
  const u32 N8 = (BT * Hh * 2) / 8;   // 8KB tile = 1024 x 8B

  const int tid = threadIdx.x, lane = tid & 31, wave = tid >> 5;
  const int gw = wave * 16;
  const int b0 = blockIdx.x * BT;

  v16bf Bi[4][4], Bh[4][4];
  float biasv[4];
#pragma unroll
  for (int g = 0; g < 4; ++g) {
    int nb = g * Hh + gw;
#pragma unroll
    for (int kt = 0; kt < 4; ++kt) {
      Bi[g][kt] = loadB_f32(Wih, Hh, nb, kt, Hh, lane);
      Bh[g][kt] = loadB_f32(Whh, Hh, nb, kt, Hh, lane);
    }
    int n = nb + (lane & 15);
    biasv[g] = bih[n] + bhh[n];
  }

  for (int i = tid; i < BT * Hh; i += NTH) hs[i] = (bf16)0.0f;

  v8f cst[2], hv[2];
#pragma unroll
  for (int m = 0; m < 2; ++m)
#pragma unroll
    for (int e = 0; e < 8; ++e) { cst[m][e] = 0.0f; hv[m][e] = 0.0f; }

  if (wave == 0)  // prefetch h1[0] tile into buffer 0 (LDS offset 8192)
    tdm_load((u32)(BT * Hh * 2), hin + ((size_t)0 * Bb + b0) * Hh, N8);
  __syncthreads();

#pragma unroll 1
  for (int t = 0; t < Tt; ++t) {
    // completes: input prefetch for this step + prior h store (before rewrite)
    if (wave == 0) wait_tensor0();
    __syncthreads();

    if (t > 0) {
#pragma unroll
      for (int m = 0; m < 2; ++m)
#pragma unroll
        for (int e = 0; e < 8; ++e) {
          int row = m * 16 + e + ((lane >> 4) & 1) * 8;
          hs[row * Hh + gw + (lane & 15)] = (bf16)hv[m][e];
        }
    }
    __syncthreads();

    if (wave == 0) {
      if (t > 0)
        tdm_store(0u, hout + ((size_t)(t - 1) * Bb + b0) * Hh, N8);
      if (t + 1 < Tt) {  // prefetch next input tile into the other buffer
        u32 off_next = ((t + 1) & 1) ? (u32)(BT * Hh * 4) : (u32)(BT * Hh * 2);
        tdm_load(off_next, hin + ((size_t)(t + 1) * Bb + b0) * Hh, N8);
      }
    }

    // current input buffer (no pointer-array: keep it pure address math)
    const bf16* is_ = smem + BT * Hh + (t & 1) * (BT * Hh);

    v16bf Ai[2][4], Ah[2][4];
#pragma unroll
    for (int m = 0; m < 2; ++m)
#pragma unroll
      for (int kt = 0; kt < 4; ++kt) {
        Ai[m][kt] = loadA_bf16(is_, Hh, m, kt, lane);
        Ah[m][kt] = loadA_bf16(hs, Hh, m, kt, lane);
      }

    v8f gacc[4][2];
#pragma unroll
    for (int g = 0; g < 4; ++g)
#pragma unroll
      for (int m = 0; m < 2; ++m) {
        v8f acc;
#pragma unroll
        for (int e = 0; e < 8; ++e) acc[e] = biasv[g];
#pragma unroll
        for (int kt = 0; kt < 4; ++kt) acc = wmma_bf16(Ai[m][kt], Bi[g][kt], acc);
#pragma unroll
        for (int kt = 0; kt < 4; ++kt) acc = wmma_bf16(Ah[m][kt], Bh[g][kt], acc);
        gacc[g][m] = acc;
      }

#pragma unroll
    for (int m = 0; m < 2; ++m)
#pragma unroll
      for (int e = 0; e < 8; ++e) {
        float ig = sigm_(gacc[0][m][e]);
        float fg = sigm_(gacc[1][m][e]);
        float gg = tanh_(gacc[2][m][e]);
        float og = sigm_(gacc[3][m][e]);
        float c = fg * cst[m][e] + ig * gg;
        cst[m][e] = c;
        hv[m][e] = og * tanh_(c);
      }
  }

  if (wave == 0) wait_tensor0();
  __syncthreads();
#pragma unroll
  for (int m = 0; m < 2; ++m)
#pragma unroll
    for (int e = 0; e < 8; ++e) {
      int row = m * 16 + e + ((lane >> 4) & 1) * 8;
      hs[row * Hh + gw + (lane & 15)] = (bf16)hv[m][e];
    }
  __syncthreads();
  if (wave == 0) {
    tdm_store(0u, hout + ((size_t)(Tt - 1) * Bb + b0) * Hh, N8);
    wait_tensor0();
  }
}

// ---------------------------------------------------------------------------
// Precompute fused head: Wc[64,128] = Wo@Wl (row-padded), bc[64] = Wo@bl + bo.
// ---------------------------------------------------------------------------
__global__ __launch_bounds__(256)
void build_head(const float* __restrict__ Wl, const float* __restrict__ bl,
                const float* __restrict__ Wo, const float* __restrict__ bo,
                bf16* __restrict__ wc, float* __restrict__ bc) {
  int tid = threadIdx.x;
  for (int idx = tid; idx < 64 * 128; idx += 256) {
    int n = idx >> 7, k = idx & 127;
    float s = 0.0f;
    if (n < 63) {
      for (int j = 0; j < 84; ++j) s += Wo[n * 84 + j] * Wl[j * 128 + k];
    }
    wc[idx] = (bf16)s;
  }
  if (tid < 64) {
    float s = 0.0f;
    if (tid < 63) {
      s = bo[tid];
      for (int j = 0; j < 84; ++j) s += Wo[tid * 84 + j] * bl[j];
    }
    bc[tid] = s;
  }
}

// ---------------------------------------------------------------------------
// Fused head GEMM: out[b,t,0:63] = h2[t,b,:] @ Wc^T + bc.
// grid = T*(B/32) = 4096, 128 threads (4 waves; wave = 16-col slice).
// ---------------------------------------------------------------------------
__global__ __launch_bounds__(128, 4)
void head_gemm(const bf16* __restrict__ h2, const bf16* __restrict__ wc,
               const float* __restrict__ bc, float* __restrict__ out) {
  const int tid = threadIdx.x, lane = tid & 31, wave = tid >> 5;
  const int bid = blockIdx.x;
  const int t = bid >> 3;
  const int b0 = (bid & 7) * BT;
  const int nc = wave * 16;

  v16bf Bw[4];
#pragma unroll
  for (int kt = 0; kt < 4; ++kt) Bw[kt] = loadB_bf16(wc, Hh, nc, kt, lane);
  const float bias = bc[nc + (lane & 15)];

  const bf16* abase = h2 + ((size_t)t * Bb + b0) * Hh;

#pragma unroll
  for (int m = 0; m < 2; ++m) {
    v8f acc;
#pragma unroll
    for (int e = 0; e < 8; ++e) acc[e] = bias;
#pragma unroll
    for (int kt = 0; kt < 4; ++kt) {
      v16bf a = loadA_bf16(abase, Hh, m, kt, lane);
      acc = wmma_bf16(a, Bw[kt], acc);
    }
    int n = nc + (lane & 15);
    if (n < 63) {
#pragma unroll
      for (int e = 0; e < 8; ++e) {
        int b = b0 + m * 16 + e + ((lane >> 4) & 1) * 8;
        out[((size_t)b * Tt + t) * 63 + n] = acc[e];
      }
    }
  }
}

// ---------------------------------------------------------------------------
extern "C" void kernel_launch(void* const* d_in, const int* in_sizes, int n_in,
                              void* d_out, int out_size, void* d_ws, size_t ws_size,
                              hipStream_t stream) {
  const float* x    = (const float*)d_in[0];
  const float* Wih0 = (const float*)d_in[1];
  const float* Whh0 = (const float*)d_in[2];
  const float* bih0 = (const float*)d_in[3];
  const float* bhh0 = (const float*)d_in[4];
  const float* Wih1 = (const float*)d_in[5];
  const float* Whh1 = (const float*)d_in[6];
  const float* bih1 = (const float*)d_in[7];
  const float* bhh1 = (const float*)d_in[8];
  const float* Wl   = (const float*)d_in[9];
  const float* bl   = (const float*)d_in[10];
  const float* Wo   = (const float*)d_in[11];
  const float* bo   = (const float*)d_in[12];

  const size_t hbytes = (size_t)Tt * Bb * Hh * sizeof(bf16);  // 33.5 MB
  char* ws = (char*)d_ws;
  bf16*  h1 = (bf16*)ws;
  bf16*  h2 = (bf16*)(ws + hbytes);
  bf16*  wc = (bf16*)(ws + 2 * hbytes);
  float* bc = (float*)(ws + 2 * hbytes + 64 * 128 * sizeof(bf16));

  build_head<<<1, 256, 0, stream>>>(Wl, bl, Wo, bo, wc, bc);
  lstm_layer0<<<Bb / BT, NTH, 0, stream>>>(x, Wih0, Whh0, bih0, bhh0, h1);
  lstm_layer1<<<Bb / BT, NTH, 0, stream>>>(h1, Wih1, Whh1, bih1, bhh1, h2);
  head_gemm<<<Tt * (Bb / BT), 128, 0, stream>>>(h2, wc, bc, (float*)d_out);
}
